// TemporalModule_6339371729605
// MI455X (gfx1250) — matrix operand order repaired and checked by reading, not compiled, not run on hardware
//
#include <hip/hip_runtime.h>

#define BDIM 8
#define TDIM 1024
#define DDIM 512
#define HDIM 8
#define FDIM 64
#define KSEL 307            // int(0.3 * 1024)
#define ALPHA  0.2f
#define GAMMA  0.0625f
#define BIASC  0.2f
#define LN_EPS 1e-5f
#define NEG_INF (-9e15f)
#define BANDR 44            // exp underflows (f32) for |dt|>=41; margin to 44
#define SWIN  (2*BANDR+1)   // 89
#define SPAD  96

typedef __attribute__((ext_vector_type(16))) _Float16 v16h;
typedef __attribute__((ext_vector_type(8)))  _Float16 v8h;
typedef __attribute__((ext_vector_type(8)))  float    v8f;
typedef int v4i_b128 __attribute__((vector_size(16)));   // matches async-LDS builtin param

#if __has_builtin(__builtin_amdgcn_global_load_async_to_lds_b128) && __has_builtin(__builtin_amdgcn_s_wait_asynccnt)
#define HAVE_ASYNC_LDS 1
#else
#define HAVE_ASYNC_LDS 0
#endif

static __device__ inline v16h cat8(v8h lo, v8h hi) {
  return __builtin_shufflevector(lo, hi, 0,1,2,3,4,5,6,7,8,9,10,11,12,13,14,15);
}
// A fragment: 16(M) x 32(K) f16 tile in LDS, row stride 32 halves.
// A layout (contiguous K, matching B): lanes 0-15 = rows, K 0..15; lanes 16-31: K 16..31.
static __device__ inline v16h ld_fragA(const _Float16* base, int lane) {
  const _Float16* p = base + (lane & 15) * 32 + ((lane >> 4) << 4);
  return cat8(*(const v8h*)p, *(const v8h*)(p + 8));
}
// B fragment from K-transposed LDS tile: [16 n rows][stride halves], contiguous K16 per lane half.
static __device__ inline v16h ld_fragB_s(const _Float16* base, int lane, int stride) {
  const _Float16* p = base + (lane & 15) * stride + ((lane >> 4) << 4);
  return cat8(*(const v8h*)p, *(const v8h*)(p + 8));
}

// ---- async (direct-to-LDS) A-tile staging: 64 rows x 32 halves, b128 chunks
static __device__ inline void stage_A(_Float16* dstLds, const _Float16* srcGlb,
                                      int lda, int kb, int tid) {
#if HAVE_ASYNC_LDS
  for (int c = tid; c < 256; c += 128) {
    int r = c >> 2, cc = (c & 3) * 8;
    __builtin_amdgcn_global_load_async_to_lds_b128(
        (__attribute__((address_space(1))) v4i_b128*)(srcGlb + (size_t)r * lda + kb + cc),
        (__attribute__((address_space(3))) v4i_b128*)(dstLds + r * 32 + cc),
        0, 0);
  }
#else
  for (int c = tid; c < 256; c += 128) {
    int r = c >> 2, cc = (c & 3) * 8;
    *(v8h*)&dstLds[r * 32 + cc] = *(const v8h*)&srcGlb[(size_t)r * lda + kb + cc];
  }
#endif
}
static __device__ inline void wait_async_then_barrier() {
#if HAVE_ASYNC_LDS
  __builtin_amdgcn_s_wait_asynccnt(0);
#endif
  __syncthreads();
}
// stage Bt[n*256 + k] = Bsrc[k*ldb + n] for 64 n x 256 k (vector read over n)
static __device__ inline void stage_Bt(_Float16* Bt, const _Float16* Bsrc, int ldb, int tid) {
  for (int c = tid; c < 2048; c += 128) {
    int k = c >> 3, ng = (c & 7) * 8;
    v8h v = *(const v8h*)&Bsrc[(size_t)k * ldb + ng];
#pragma unroll
    for (int j = 0; j < 8; ++j) Bt[(ng + j) * 256 + k] = v[j];
  }
}

// ---------------------------------------------------------------- converts
__global__ void k_f32_to_f16(const float* __restrict__ src, _Float16* __restrict__ dst, int n) {
  int i = blockIdx.x * blockDim.x + threadIdx.x;
  if (i < n) dst[i] = (_Float16)src[i];
}

// ------------------------------------------------------------ row L2 norms
__global__ __launch_bounds__(128) void k_rownorm(const float* __restrict__ x, float* __restrict__ mags) {
  __shared__ float red[128];
  int row = blockIdx.x;                         // b*T + t
  const float* p = x + (size_t)row * DDIM;
  float s = 0.f;
  for (int i = threadIdx.x; i < DDIM; i += 128) { float v = p[i]; s += v * v; }
  red[threadIdx.x] = s; __syncthreads();
  for (int off = 64; off > 0; off >>= 1) { if (threadIdx.x < off) red[threadIdx.x] += red[threadIdx.x + off]; __syncthreads(); }
  if (threadIdx.x == 0) mags[row] = sqrtf(red[0]);
}

// ------------------------------------------- exact top-k via stable rank
__global__ void k_select(const float* __restrict__ mags, int* __restrict__ sel) {
  int idx = blockIdx.x * blockDim.x + threadIdx.x;       // b*T + t
  if (idx >= BDIM * TDIM) return;
  int b = idx >> 10, t = idx & 1023;
  const float* m = mags + b * TDIM;
  float mt = m[t];
  int rank = 0;
  for (int j = 0; j < TDIM; ++j) {
    float mj = m[j];
    rank += (mj > mt) || (mj == mt && j < t);
  }
  sel[idx] = (rank < KSEL) ? 1 : 0;
}

// ------------------------------- WMMA GEMM: Wh[b,h] = x[b](TxD) @ W[h](DxF)
__global__ __launch_bounds__(128) void k_gemm_wh(const _Float16* __restrict__ xh,
                                                 const _Float16* __restrict__ W16,
                                                 float* __restrict__ Wh) {
  __shared__ __align__(16) _Float16 Bt[64 * 256];      // persistent half-K B, 32KB
  __shared__ __align__(16) _Float16 As[2][64 * 32];    // double-buffered A, 8KB
  int t0 = blockIdx.x * 64, h = blockIdx.y, b = blockIdx.z;
  int tid = threadIdx.x, wave = tid >> 5, lane = tid & 31;
  const _Float16* Ab = xh + ((size_t)b * TDIM + t0) * DDIM;
  const _Float16* Bb = W16 + (size_t)h * DDIM * FDIM;
  v8f acc[4] = {};
  for (int half = 0; half < 2; ++half) {
    int k0 = half * 256;
    stage_Bt(Bt, Bb + (size_t)k0 * FDIM, FDIM, tid);
    stage_A(As[0], Ab, DDIM, k0, tid);
    wait_async_then_barrier();
    for (int kb8 = 0; kb8 < 8; ++kb8) {
      if (kb8 < 7) stage_A(As[(kb8 + 1) & 1], Ab, DDIM, k0 + (kb8 + 1) * 32, tid);
      v16h a = ld_fragA(&As[kb8 & 1][(wave * 16) * 32], lane);
#pragma unroll
      for (int nt = 0; nt < 4; ++nt) {
        v16h bf = ld_fragB_s(&Bt[(nt * 16) * 256 + kb8 * 32], lane, 256);
        acc[nt] = __builtin_amdgcn_wmma_f32_16x16x32_f16(false, a, false, bf, (short)0, acc[nt], false, false);
      }
      wait_async_then_barrier();
    }
  }
  float* Cb = Wh + (((size_t)b * HDIM + h) * TDIM + t0 + wave * 16) * FDIM;
  int col = lane & 15, rbase = (lane >> 4) * 8;
#pragma unroll
  for (int nt = 0; nt < 4; ++nt)
#pragma unroll
    for (int r = 0; r < 8; ++r)
      Cb[(size_t)(rbase + r) * FDIM + nt * 16 + col] = acc[nt][r];
}

// ------------------------------------------------- e_src / e_dst dot prods
__global__ void k_edge(const float* __restrict__ Wh, const float* __restrict__ a_src,
                       const float* __restrict__ a_dst, float* __restrict__ es, float* __restrict__ ed) {
  int idx = blockIdx.x * blockDim.x + threadIdx.x;   // (b*H + h)*T + t
  if (idx >= BDIM * HDIM * TDIM) return;
  int h = (idx >> 10) & (HDIM - 1);
  const float* w = Wh + (size_t)idx * FDIM;
  const float* as = a_src + h * FDIM;
  const float* ad = a_dst + h * FDIM;
  float s = 0.f, d = 0.f;
  for (int f = 0; f < FDIM; ++f) { float v = w[f]; s = fmaf(v, as[f], s); d = fmaf(v, ad[f], d); }
  es[idx] = s; ed[idx] = d;
}

// ---------------------------------------- column means of Wh (uniform rows)
__global__ void k_whmean(const float* __restrict__ Wh, float* __restrict__ wm) {
  int idx = blockIdx.x * blockDim.x + threadIdx.x;   // bh*F + f
  if (idx >= BDIM * HDIM * FDIM) return;
  int f = idx & (FDIM - 1), bh = idx >> 6;
  const float* p = Wh + (size_t)bh * TDIM * FDIM + f;
  float s = 0.f;
  for (int t = 0; t < TDIM; ++t) s += p[(size_t)t * FDIM];
  wm[idx] = s * (1.0f / TDIM);
}

// ---- banded masked softmax + band@Wh + ELU + signed-sqrt, one wave per row
__global__ __launch_bounds__(256) void k_attn(const float* __restrict__ Wh, const float* __restrict__ es,
                                              const float* __restrict__ ed, const int* __restrict__ sel,
                                              const float* __restrict__ wm, float* __restrict__ tmp) {
  __shared__ float pbuf[8][SPAD];
  int wave = threadIdx.x >> 5, lane = threadIdx.x & 31;
  int g = blockIdx.x * 8 + wave;                    // global row: (b*H+h)*T + t
  int bh = g >> 10, t = g & 1023;
  int b = bh >> 3, h = bh & 7;
  const int* selb = sel + b * TDIM;
  int selt = selb[t];
  float est = es[(size_t)bh * TDIM + t];
  const float* edb = ed + (size_t)bh * TDIM;

  float lv[3]; float lmax = NEG_INF;
#pragma unroll
  for (int i = 0; i < 3; ++i) {
    int sl = lane + i * 32;
    int s = t - BANDR + sl;
    float l = NEG_INF;
    if (sl < SWIN && s >= 0 && s < TDIM) {
      float dd = (float)(t - s);
      float a = expf(-GAMMA * dd * dd - BIASC);     // ==0 (f32 underflow) outside band
      if (a > 0.f && (selt || selb[s])) {
        float e = est + edb[s];
        l = (e > 0.f) ? e : ALPHA * e;
      }
    }
    lv[i] = l; lmax = fmaxf(lmax, l);
  }
  for (int off = 16; off > 0; off >>= 1) lmax = fmaxf(lmax, __shfl_xor(lmax, off, 32));
  bool uniform = (lmax <= -8.9e15f);                // all logits masked -> softmax uniform 1/T
  float psum = 0.f;
#pragma unroll
  for (int i = 0; i < 3; ++i) {
    int sl = lane + i * 32;
    float p = 0.f;
    if (!uniform && sl < SWIN && lv[i] > -8.9e15f) p = expf(lv[i] - lmax);
    pbuf[wave][sl] = p;
    psum += p;
  }
  for (int off = 16; off > 0; off >>= 1) psum += __shfl_xor(psum, off, 32);

  const float* Whb = Wh + (size_t)bh * TDIM * FDIM;
  float acc0, acc1;
  if (uniform) {
    acc0 = wm[bh * FDIM + lane];
    acc1 = wm[bh * FDIM + lane + 32];
  } else {
    acc0 = 0.f; acc1 = 0.f;
    int slo = (t - BANDR > 0) ? t - BANDR : 0;
    int shi = (t + BANDR < TDIM - 1) ? t + BANDR : TDIM - 1;
    for (int s = slo; s <= shi; ++s) {
      float p = pbuf[wave][s - (t - BANDR)];
      const float* wr = Whb + (size_t)s * FDIM;
      acc0 = fmaf(p, wr[lane], acc0);
      acc1 = fmaf(p, wr[lane + 32], acc1);
    }
    float inv = 1.0f / psum;
    acc0 *= inv; acc1 *= inv;
  }
  float v0 = (acc0 > 0.f) ? acc0 : (expf(acc0) - 1.f);   // ELU
  float v1 = (acc1 > 0.f) ? acc1 : (expf(acc1) - 1.f);
  v0 = (v0 >= 0.f) ? sqrtf(v0) : -sqrtf(-v0);            // signed sqrt
  v1 = (v1 >= 0.f) ? sqrtf(v1) : -sqrtf(-v1);
  float* trow = tmp + ((size_t)b * TDIM + t) * DDIM + h * FDIM;
  trow[lane] = v0; trow[lane + 32] = v1;
}

// ------------------------------------------ L2 norm over T per (b, channel)
__global__ void k_colnorm(const float* __restrict__ tmp, float* __restrict__ cn) {
  int idx = blockIdx.x * blockDim.x + threadIdx.x;   // b*D + dd
  if (idx >= BDIM * DDIM) return;
  int b = idx / DDIM, dd = idx % DDIM;
  const float* p = tmp + (size_t)b * TDIM * DDIM + dd;
  float s = 0.f;
  for (int t = 0; t < TDIM; ++t) { float v = p[(size_t)t * DDIM]; s = fmaf(v, v, s); }
  cn[idx] = fmaxf(sqrtf(s), 1e-12f);
}

__global__ void k_normalize_h(const float* __restrict__ tmp, const float* __restrict__ cn,
                              _Float16* __restrict__ outh) {
  int idx = blockIdx.x * blockDim.x + threadIdx.x;   // b*T*D flat
  if (idx >= BDIM * TDIM * DDIM) return;
  int dd = idx & (DDIM - 1);
  int b = idx / (TDIM * DDIM);
  outh[idx] = (_Float16)(tmp[idx] / cn[b * DDIM + dd]);
}

// ------------------ WMMA GEMM: y0 = x + tmpn(8192x512) @ W2(512x512) + b2
__global__ __launch_bounds__(128) void k_gemm2(const _Float16* __restrict__ Ah,
                                               const _Float16* __restrict__ Bh,
                                               const float* __restrict__ x,
                                               const float* __restrict__ b2,
                                               float* __restrict__ y0) {
  __shared__ __align__(16) _Float16 Bt[64 * 256];
  __shared__ __align__(16) _Float16 As[2][64 * 32];
  int m0 = blockIdx.x * 64, n0b = blockIdx.y * 64;
  int tid = threadIdx.x, wave = tid >> 5, lane = tid & 31;
  const _Float16* Ab = Ah + (size_t)m0 * DDIM;
  v8f acc[4] = {};
  for (int half = 0; half < 2; ++half) {
    int k0 = half * 256;
    stage_Bt(Bt, Bh + (size_t)k0 * DDIM + n0b, DDIM, tid);
    stage_A(As[0], Ab, DDIM, k0, tid);
    wait_async_then_barrier();
    for (int kb8 = 0; kb8 < 8; ++kb8) {
      if (kb8 < 7) stage_A(As[(kb8 + 1) & 1], Ab, DDIM, k0 + (kb8 + 1) * 32, tid);
      v16h a = ld_fragA(&As[kb8 & 1][(wave * 16) * 32], lane);
#pragma unroll
      for (int nt = 0; nt < 4; ++nt) {
        v16h bf = ld_fragB_s(&Bt[(nt * 16) * 256 + kb8 * 32], lane, 256);
        acc[nt] = __builtin_amdgcn_wmma_f32_16x16x32_f16(false, a, false, bf, (short)0, acc[nt], false, false);
      }
      wait_async_then_barrier();
    }
  }
  int col = lane & 15, rbase = (lane >> 4) * 8;
#pragma unroll
  for (int nt = 0; nt < 4; ++nt)
#pragma unroll
    for (int r = 0; r < 8; ++r) {
      size_t m = (size_t)m0 + wave * 16 + rbase + r;
      int n = n0b + nt * 16 + col;
      y0[m * DDIM + n] = acc[nt][r] + x[m * DDIM + n] + b2[n];
    }
}

// ----------------------------------- LayerNorm over D + transpose to [B,D,T]
__global__ __launch_bounds__(256) void k_ln_out(const float* __restrict__ y0, const float* __restrict__ g,
                                                const float* __restrict__ be, float* __restrict__ out) {
  __shared__ float red[256];
  int row = blockIdx.x;                       // b*T + t
  int b = row >> 10, t = row & 1023;
  const float* y = y0 + (size_t)row * DDIM;
  int i = threadIdx.x;
  float v0 = y[i], v1 = y[i + 256];
  red[i] = v0 + v1; __syncthreads();
  for (int off = 128; off > 0; off >>= 1) { if (i < off) red[i] += red[i + off]; __syncthreads(); }
  float mu = red[0] * (1.0f / DDIM);
  __syncthreads();
  float d0 = v0 - mu, d1 = v1 - mu;
  red[i] = d0 * d0 + d1 * d1; __syncthreads();
  for (int off = 128; off > 0; off >>= 1) { if (i < off) red[i] += red[i + off]; __syncthreads(); }
  float rstd = rsqrtf(red[0] * (1.0f / DDIM) + LN_EPS);
  out[((size_t)b * DDIM + i) * TDIM + t]       = d0 * rstd * g[i] + be[i];
  out[((size_t)b * DDIM + i + 256) * TDIM + t] = d1 * rstd * g[i + 256] + be[i + 256];
}

// ---------------------------------------------------------------- launcher
extern "C" void kernel_launch(void* const* d_in, const int* in_sizes, int n_in,
                              void* d_out, int out_size, void* d_ws, size_t ws_size,
                              hipStream_t stream) {
  const float* x     = (const float*)d_in[0];
  const float* W     = (const float*)d_in[1];
  const float* a_src = (const float*)d_in[2];
  const float* a_dst = (const float*)d_in[3];
  const float* W2    = (const float*)d_in[4];
  const float* b2    = (const float*)d_in[5];
  const float* ln_g  = (const float*)d_in[6];
  const float* ln_b  = (const float*)d_in[7];
  float* out = (float*)d_out;

  char* ws = (char*)d_ws;
  size_t off = 0;
  auto alloc = [&](size_t bytes) { void* p = ws + off; off += (bytes + 255) & ~(size_t)255; return p; };
  const size_t NX  = (size_t)BDIM * TDIM * DDIM;       // 4,194,304
  const size_t NW  = (size_t)HDIM * DDIM * FDIM;       //   262,144
  const size_t NW2 = (size_t)DDIM * DDIM;              //   262,144
  _Float16* xh   = (_Float16*)alloc(NX * 2);
  _Float16* W16  = (_Float16*)alloc(NW * 2);
  _Float16* W2h  = (_Float16*)alloc(NW2 * 2);
  float*    Wh   = (float*)alloc((size_t)BDIM * HDIM * TDIM * FDIM * 4);
  float*    mags = (float*)alloc((size_t)BDIM * TDIM * 4);
  int*      sel  = (int*)alloc((size_t)BDIM * TDIM * 4);
  float*    es   = (float*)alloc((size_t)BDIM * HDIM * TDIM * 4);
  float*    ed   = (float*)alloc((size_t)BDIM * HDIM * TDIM * 4);
  float*    wm   = (float*)alloc((size_t)BDIM * HDIM * FDIM * 4);
  float*    tmp  = (float*)alloc(NX * 4);
  float*    cn   = (float*)alloc((size_t)BDIM * DDIM * 4);
  _Float16* tmpn = (_Float16*)alloc(NX * 2);
  float*    y0   = (float*)alloc(NX * 4);

  k_f32_to_f16<<<(int)((NX  + 255) / 256), 256, 0, stream>>>(x,  xh,  (int)NX);
  k_f32_to_f16<<<(int)((NW  + 255) / 256), 256, 0, stream>>>(W,  W16, (int)NW);
  k_f32_to_f16<<<(int)((NW2 + 255) / 256), 256, 0, stream>>>(W2, W2h, (int)NW2);

  k_rownorm<<<BDIM * TDIM, 128, 0, stream>>>(x, mags);
  k_select<<<(BDIM * TDIM + 255) / 256, 256, 0, stream>>>(mags, sel);

  k_gemm_wh<<<dim3(TDIM / 64, HDIM, BDIM), 128, 0, stream>>>(xh, W16, Wh);

  k_edge<<<(BDIM * HDIM * TDIM + 255) / 256, 256, 0, stream>>>(Wh, a_src, a_dst, es, ed);
  k_whmean<<<(BDIM * HDIM * FDIM + 255) / 256, 256, 0, stream>>>(Wh, wm);

  k_attn<<<BDIM * HDIM * TDIM / 8, 256, 0, stream>>>(Wh, es, ed, sel, wm, tmp);

  k_colnorm<<<(BDIM * DDIM + 255) / 256, 256, 0, stream>>>(tmp, cn);
  k_normalize_h<<<(int)((NX + 255) / 256), 256, 0, stream>>>(tmp, cn, tmpn);

  k_gemm2<<<dim3(BDIM * TDIM / 64, DDIM / 64), 128, 0, stream>>>(tmpn, W2h, x, b2, y0);

  k_ln_out<<<BDIM * TDIM, 256, 0, stream>>>(y0, ln_g, ln_b, out);
}
